// SAModule_60043642798272
// MI455X (gfx1250) — matrix operand order repaired and checked by reading, not compiled, not run on hardware
//
#include <hip/hip_runtime.h>

// ---------------------------------------------------------------------------
// Problem constants (from the reference)
// ---------------------------------------------------------------------------
#define NPTS   32768   // N input points
#define FCH    64      // feature channels
#define MSEL   8192    // FPS-selected centroids
#define KNB    64      // max neighbors
#define HCH    128     // hidden
#define OCH    128     // out channels
#define FPAD   96      // 67 padded to 3*32 for f16 WMMA k-steps
#define CAP    256     // candidate capacity per centroid
#define CPB    16      // centroids per block in the MLP kernel

typedef _Float16 half8 __attribute__((ext_vector_type(8)));
typedef _Float16 v16h  __attribute__((ext_vector_type(16)));
typedef float    v8f   __attribute__((ext_vector_type(8)));

static __device__ inline v8f wmma_f16(v16h a, v16h b, v8f c) {
  return __builtin_amdgcn_wmma_f32_16x16x32_f16(
      /*neg_a=*/false, a, /*neg_b=*/false, b,
      /*c_mod=*/(short)0, c, /*reuse_a=*/false, /*reuse_b=*/false);
}

// B fragment (32x16, 16-bit): lane n = l&15; lanes<16 load K = kb..kb+15 (this code's order, as ldA),
// lanes>=16 load K = kb+16..kb+31 -> 16 contiguous halfs.
static __device__ inline v16h ldB(const _Float16* p) {
  v16h r;
  ((half8*)&r)[0] = *(const half8*)(p);
  ((half8*)&r)[1] = *(const half8*)(p + 8);
  return r;
}

// A fragment (16x32, 16-bit): lane row = l&15; lane<16: K kb..kb+15,
// lane>=16: K kb+16..kb+31.
static __device__ inline v16h ldA(const _Float16* rowp, int kchunk, bool hi) {
  const _Float16* p = rowp + kchunk * 32 + (hi ? 16 : 0);
  v16h r;
  ((half8*)&r)[0] = *(const half8*)(p);
  ((half8*)&r)[1] = *(const half8*)(p + 8);
  return r;
}

// ---------------------------------------------------------------------------
// Kernel 0: weight prep (fp32 -> f16, transposed) + zero the sel_batch tail
// ---------------------------------------------------------------------------
__global__ void prep_kernel(const float* __restrict__ W1, const float* __restrict__ W2,
                            _Float16* __restrict__ W1t, _Float16* __restrict__ W2t,
                            float* __restrict__ tail, int tailN) {
  int i = blockIdx.x * 256 + threadIdx.x;
  if (i < HCH * FPAD) {                       // W1t[h][f] = W1[f][h], pad f>=67 with 0
    int h = i / FPAD, f = i % FPAD;
    W1t[i] = (_Float16)((f < FCH + 3) ? W1[f * HCH + h] : 0.0f);
  } else if (i < HCH * FPAD + OCH * HCH) {    // W2t[o][h] = W2[h][o]
    int j = i - HCH * FPAD;
    int o = j / HCH, h = j % HCH;
    W2t[j] = (_Float16)W2[h * OCH + o];
  } else {
    int j = i - (HCH * FPAD + OCH * HCH);
    if (j < tailN) tail[j] = 0.0f;            // sel_batch == zeros
  }
}

// ---------------------------------------------------------------------------
// Kernel 1: farthest point sampling — one persistent 1024-thread workgroup.
// min_d kept in registers (32 per thread, strided ownership), argmax via
// wave shuffle + LDS tree, tie-break = smallest index (matches jnp.argmax).
// ---------------------------------------------------------------------------
__global__ __launch_bounds__(1024) void fps_kernel(const float* __restrict__ pos,
                                                   int* __restrict__ sel,
                                                   float* __restrict__ sel_pos) {
  __shared__ float s_val[32];
  __shared__ int   s_idx[32];
  __shared__ float s_cur[3];

  const int t = threadIdx.x;
  const int lane = t & 31, wv = t >> 5;

  float cx = pos[0], cy = pos[1], cz = pos[2];
  float mind[32];
#pragma unroll
  for (int j = 0; j < 32; ++j) {
    int p = t + j * 1024;
    float dx = pos[3 * p] - cx, dy = pos[3 * p + 1] - cy, dz = pos[3 * p + 2] - cz;
    mind[j] = dx * dx + dy * dy + dz * dz;
  }
  if (t == 0) {
    sel[0] = 0;
    sel_pos[0] = cx; sel_pos[1] = cy; sel_pos[2] = cz;
  }

  for (int i = 1; i < MSEL; ++i) {
    // local argmax (ascending p order -> first-max kept)
    float bv = -1.0f; int bi = t;
#pragma unroll
    for (int j = 0; j < 32; ++j) {
      int p = t + j * 1024;
      if (mind[j] > bv) { bv = mind[j]; bi = p; }
    }
    // wave reduce (max, smallest index on tie)
#pragma unroll
    for (int off = 16; off; off >>= 1) {
      float ov = __shfl_xor(bv, off, 32);
      int   oi = __shfl_xor(bi, off, 32);
      if (ov > bv || (ov == bv && oi < bi)) { bv = ov; bi = oi; }
    }
    if (lane == 0) { s_val[wv] = bv; s_idx[wv] = bi; }
    __syncthreads();
    if (wv == 0) {
      bv = s_val[lane]; bi = s_idx[lane];
#pragma unroll
      for (int off = 16; off; off >>= 1) {
        float ov = __shfl_xor(bv, off, 32);
        int   oi = __shfl_xor(bi, off, 32);
        if (ov > bv || (ov == bv && oi < bi)) { bv = ov; bi = oi; }
      }
      if (lane == 0) {
        float nx = pos[3 * bi], ny = pos[3 * bi + 1], nz = pos[3 * bi + 2];
        s_cur[0] = nx; s_cur[1] = ny; s_cur[2] = nz;
        sel[i] = bi;
        sel_pos[3 * i] = nx; sel_pos[3 * i + 1] = ny; sel_pos[3 * i + 2] = nz;
      }
    }
    __syncthreads();
    cx = s_cur[0]; cy = s_cur[1]; cz = s_cur[2];
#pragma unroll
    for (int j = 0; j < 32; ++j) {
      int p = t + j * 1024;
      float dx = pos[3 * p] - cx, dy = pos[3 * p + 1] - cy, dz = pos[3 * p + 2] - cz;
      float d = dx * dx + dy * dy + dz * dz;
      mind[j] = fminf(mind[j], d);
    }
  }
}

// ---------------------------------------------------------------------------
// Kernel 2: radius ball query, nearest-first, capped at K=64.
// One wave per centroid; candidates (d2 <= r^2) gathered into LDS, then
// iterative wave-parallel argmin selection. Unused slots: idx=0, mask=0
// (identical output under the masked max).
// ---------------------------------------------------------------------------
__global__ __launch_bounds__(256) void knn_kernel(const float* __restrict__ pos,
                                                  const float* __restrict__ sel_pos,
                                                  int* __restrict__ nbr,
                                                  int* __restrict__ msk) {
  __shared__ float cd[8][CAP];
  __shared__ int   ci[8][CAP];
  __shared__ int   cnt[8];

  const int t = threadIdx.x, lane = t & 31, wv = t >> 5;
  const int m = blockIdx.x * 8 + wv;
  if (lane == 0) cnt[wv] = 0;

  const float qx = sel_pos[3 * m], qy = sel_pos[3 * m + 1], qz = sel_pos[3 * m + 2];
  const float r2 = 0.08f * 0.08f;

  for (int p = lane; p < NPTS; p += 32) {
    float dx = pos[3 * p] - qx, dy = pos[3 * p + 1] - qy, dz = pos[3 * p + 2] - qz;
    float d = dx * dx + dy * dy + dz * dz;
    if (d <= r2) {
      int slot = atomicAdd(&cnt[wv], 1);
      if (slot < CAP) { cd[wv][slot] = d; ci[wv][slot] = p; }
    }
  }
  __syncthreads();

  const int n = min(cnt[wv], CAP);
  for (int s = 0; s < KNB; ++s) {
    float bd = 3.0e37f; int bi = 0x7fffffff; int bs = -1;
    for (int j = lane; j < n; j += 32) {
      float d = cd[wv][j]; int id = ci[wv][j];
      if (d < bd || (d == bd && id < bi)) { bd = d; bi = id; bs = j; }
    }
#pragma unroll
    for (int off = 16; off; off >>= 1) {
      float od = __shfl_xor(bd, off, 32);
      int   oi = __shfl_xor(bi, off, 32);
      int   os = __shfl_xor(bs, off, 32);
      if (od < bd || (od == bd && oi < bi)) { bd = od; bi = oi; bs = os; }
    }
    if (lane == 0) {
      if (s < n) {
        nbr[m * KNB + s] = bi;
        msk[m * KNB + s] = 1;
        cd[wv][bs] = 3.0e38f;   // mark consumed
      } else {
        nbr[m * KNB + s] = 0;
        msk[m * KNB + s] = 0;
      }
    }
  }
}

// ---------------------------------------------------------------------------
// Kernel 3: PointConv MLP + masked max — the WMMA workhorse.
// Block = 256 threads = 8 waves; wave w owns output channels [16w, 16w+16).
// Per centroid: feat(64x96,f16) in LDS; GEMM1 64x96x128 (3 k-steps), ReLU,
// h(64x128,f16) in LDS; GEMM2 64x128x128 (4 k-steps); masked max over the
// 64 neighbor rows stays within one wave + one shfl_xor(16).
// ---------------------------------------------------------------------------
__global__ __launch_bounds__(256) void mlp_kernel(const float* __restrict__ x,
                                                  const float* __restrict__ pos,
                                                  const float* __restrict__ sel_pos,
                                                  const float* __restrict__ b1,
                                                  const float* __restrict__ b2,
                                                  const int* __restrict__ nbr,
                                                  const int* __restrict__ msk,
                                                  const _Float16* __restrict__ W1t_g,
                                                  const _Float16* __restrict__ W2t_g,
                                                  float* __restrict__ out) {
  extern __shared__ char smem[];
  _Float16* sW1 = (_Float16*)smem;              // 128 x 96   (24576 B)
  _Float16* sW2 = sW1 + HCH * FPAD;             // 128 x 128  (32768 B)
  _Float16* sF  = sW2 + OCH * HCH;              // 64  x 96   (12288 B)
  _Float16* sH  = sF + KNB * FPAD;              // 64  x 128  (16384 B)
  int* sMask    = (int*)(sH + KNB * HCH);       // 64         (256 B)

  const int t = threadIdx.x, lane = t & 31, wv = t >> 5;
  const bool hi = (lane >= 16);
  const int  row = lane & 15;                   // A row / column id within tile
  const int  c = wv * 16 + row;                 // this lane's output channel

  // stage weights into LDS once (persist across CPB centroids)
  for (int i = t; i < HCH * FPAD; i += 256) sW1[i] = W1t_g[i];
  for (int i = t; i < OCH * HCH;  i += 256) sW2[i] = W2t_g[i];
  const float bias1 = b1[c];
  const float bias2 = b2[c];
  __syncthreads();

  for (int mm = 0; mm < CPB; ++mm) {
    const int m = blockIdx.x * CPB + mm;

    // --- build feat tile: [n][col] = concat(x_j, pos_j - pos_i, zero-pad) ---
    {
      const int n = t >> 2, q = t & 3;          // 4 threads per neighbor row
      const int kidx = nbr[m * KNB + n];
      if (q == 0) sMask[n] = msk[m * KNB + n];
      const float sx = sel_pos[3 * m], sy = sel_pos[3 * m + 1], sz = sel_pos[3 * m + 2];
#pragma unroll
      for (int jj = 0; jj < 24; ++jj) {
        int col = q * 24 + jj;
        float v;
        if (col < FCH)            v = x[kidx * FCH + col];
        else if (col == FCH)      v = pos[3 * kidx]     - sx;
        else if (col == FCH + 1)  v = pos[3 * kidx + 1] - sy;
        else if (col == FCH + 2)  v = pos[3 * kidx + 2] - sz;
        else                      v = 0.0f;
        sF[n * FPAD + col] = (_Float16)v;
      }
    }
    __syncthreads();

    // --- GEMM1: h = relu(feat @ W1 + b1), 4 row-tiles x (this wave's col-tile) ---
    {
      v8f C[4] = {};
#pragma unroll
      for (int k = 0; k < 3; ++k) {
        v16h B = ldB(sW1 + c * FPAD + k * 32 + (hi ? 16 : 0));
#pragma unroll
        for (int rt = 0; rt < 4; ++rt) {
          v16h A = ldA(sF + (rt * 16 + row) * FPAD, k, hi);
          C[rt] = wmma_f16(A, B, C[rt]);
        }
      }
#pragma unroll
      for (int rt = 0; rt < 4; ++rt) {
#pragma unroll
        for (int v = 0; v < 8; ++v) {
          float h = C[rt][v] + bias1;
          h = fmaxf(h, 0.0f);
          int r = rt * 16 + v + (hi ? 8 : 0);   // C-layout: VGPR v -> row v (+8 hi)
          sH[r * HCH + c] = (_Float16)h;
        }
      }
    }
    __syncthreads();

    // --- GEMM2: d = h @ W2 + b2; masked max over the 64 neighbor rows ---
    {
      v8f D[4] = {};
#pragma unroll
      for (int k = 0; k < 4; ++k) {
        v16h B = ldB(sW2 + c * HCH + k * 32 + (hi ? 16 : 0));
#pragma unroll
        for (int rt = 0; rt < 4; ++rt) {
          v16h A = ldA(sH + (rt * 16 + row) * HCH, k, hi);
          D[rt] = wmma_f16(A, B, D[rt]);
        }
      }
      float mx = -__builtin_inff();
#pragma unroll
      for (int rt = 0; rt < 4; ++rt) {
#pragma unroll
        for (int v = 0; v < 8; ++v) {
          int r = rt * 16 + v + (hi ? 8 : 0);
          float val = D[rt][v] + bias2;
          if (sMask[r]) mx = fmaxf(mx, val);
        }
      }
      // lanes l and l+16 hold complementary rows of the same channel
      mx = fmaxf(mx, __shfl_xor(mx, 16, 32));
      if (!hi) out[m * OCH + c] = mx;
    }
    __syncthreads();  // protect sMask/sF/sH before next centroid's fill
  }
}

// ---------------------------------------------------------------------------
// Host-side launcher
// ---------------------------------------------------------------------------
extern "C" void kernel_launch(void* const* d_in, const int* in_sizes, int n_in,
                              void* d_out, int out_size, void* d_ws, size_t ws_size,
                              hipStream_t stream) {
  const float* x   = (const float*)d_in[0];   // [N, F]
  const float* pos = (const float*)d_in[1];   // [N, 3]
  // d_in[2] = batch (all zeros) -- unused
  const float* W1 = (const float*)d_in[3];    // [F+3, H]
  const float* b1 = (const float*)d_in[4];    // [H]
  const float* W2 = (const float*)d_in[5];    // [H, OUT]
  const float* b2 = (const float*)d_in[6];    // [OUT]

  float* out     = (float*)d_out;             // [M,128] ++ [M,3] ++ [M]
  float* sel_pos = out + (size_t)MSEL * OCH;
  float* tail    = out + (size_t)MSEL * (OCH + 3);
  int tailN      = out_size - MSEL * (OCH + 3);
  if (tailN < 0) tailN = 0;

  // workspace layout
  char* ws = (char*)d_ws;
  int*      sel = (int*)(ws);
  int*      nbr = (int*)(ws + sizeof(int) * MSEL);
  int*      msk = (int*)(ws + sizeof(int) * (MSEL + MSEL * KNB));
  _Float16* W1t = (_Float16*)(ws + sizeof(int) * (MSEL + 2 * MSEL * KNB));
  _Float16* W2t = W1t + HCH * FPAD;

  // 0) weight transpose/convert + zero sel_batch tail
  {
    int total = HCH * FPAD + OCH * HCH + tailN;
    prep_kernel<<<(total + 255) / 256, 256, 0, stream>>>(W1, W2, W1t, W2t, tail, tailN);
  }
  // 1) farthest point sampling (single persistent workgroup)
  fps_kernel<<<1, 1024, 0, stream>>>(pos, sel, sel_pos);
  // 2) radius ball query (1 wave per centroid)
  knn_kernel<<<MSEL / 8, 256, 0, stream>>>(pos, sel_pos, nbr, msk);
  // 3) PointConv MLP + masked max (WMMA)
  {
    size_t shb = sizeof(_Float16) * (HCH * FPAD + OCH * HCH + KNB * FPAD + KNB * HCH)
               + sizeof(int) * KNB;
    mlp_kernel<<<MSEL / CPB, 256, shb, stream>>>(x, pos, sel_pos, b1, b2,
                                                 nbr, msk, W1t, W2t, out);
  }
}
